// ImprovedPointNet2Seg_5583457485357
// MI455X (gfx1250) — matrix operand order repaired and checked by reading, not compiled, not run on hardware
//
#include <hip/hip_runtime.h>
#include <math.h>

typedef __attribute__((ext_vector_type(16))) _Float16 v16h;
typedef __attribute__((ext_vector_type(8)))  float    v8f;

#define BN_EPS 1e-5f

__device__ __forceinline__ int clampi(int v, int hi) { return v < hi ? v : hi; }

// =====================================================================
// GEMM via CDNA5 WMMA: Y[b,o,s] = act( sum_c W[o,c]*X[b,c,s] + bias[o] )
// One wave32 per 64(O) x 16(S) output tile: the streaming activation
// fragment (B operand) is loaded once per K-step and reused across four
// weight fragments / accumulators (4x v_wmma_f32_16x16x32_f16 per step).
// Edge tiles use clamped (always-legal) loads + select-zero, so the
// guarded path is branch-free (no per-element EXEC toggles).
// =====================================================================
__global__ __launch_bounds__(32)
void pn2_gemm_wmma(const float* __restrict__ W, const float* __restrict__ X,
                   const float* __restrict__ bias, float* __restrict__ Y,
                   int O, int C, int S, int relu) {
  const int b    = blockIdx.z;
  const int sT   = blockIdx.x * 16;
  const int oT   = blockIdx.y * 64;
  const int lane = threadIdx.x & 31;
  const int half = lane >> 4;   // lane group 0/1
  const int lm   = lane & 15;
  const float* __restrict__ Xb = X + (size_t)b * C * S;
  float* __restrict__ Yb       = Y + (size_t)b * O * S;
  const int scol  = sT + lm;
  const int scolc = clampi(scol, S - 1);       // clamped (always in-bounds)
  const bool sOK  = (scol < S);
  const bool fullO = (oT + 64 <= O);
  const bool fullS = (sT + 16 <= S);

  v8f acc[4] = {{}, {}, {}, {}};
  for (int k0 = 0; k0 < C; k0 += 32) {
    const bool fullK = (k0 + 32 <= C);
    // ---- B fragment (activations), loaded once, reused 4x -----------
    v16h bfrag;
    if (fullS && fullK) {
      if (k0 + 64 <= C)  // prefetch next K-slab of the streaming operand
        __builtin_prefetch(&Xb[(size_t)(k0 + 32 + half * 16) * S + scol], 0, 1);
#pragma unroll
      for (int e = 0; e < 16; ++e)
        bfrag[e] = (_Float16)Xb[(size_t)(k0 + half * 8 + e + (e >= 8 ? 8 : 0)) * S + scol];
    } else {
#pragma unroll
      for (int e = 0; e < 16; ++e) {
        const int k  = k0 + half * 8 + e + (e >= 8 ? 8 : 0);
        const int kc = clampi(k, C - 1);
        float bv = Xb[(size_t)kc * S + scolc];         // always legal
        bfrag[e] = (_Float16)((k < C && sOK) ? bv : 0.0f);
      }
    }
    // ---- 4 A fragments (weights, L2-hot) + 4 WMMAs ------------------
#pragma unroll
    for (int t = 0; t < 4; ++t) {
      const int orow = oT + t * 16 + lm;
      v16h afrag;
      if (fullO && fullK) {
#pragma unroll
        for (int e = 0; e < 16; ++e) {
          const int v    = e >> 1;
          const int lohi = e & 1;
          const int kb   = (v < 4) ? (v * 2 + half * 8) : (16 + (v - 4) * 2 + half * 8);
          afrag[e] = (_Float16)W[(size_t)orow * C + (k0 + kb + lohi)];
        }
      } else {
        const int oc = clampi(orow, O - 1);
#pragma unroll
        for (int e = 0; e < 16; ++e) {
          const int v    = e >> 1;
          const int lohi = e & 1;
          const int kb   = (v < 4) ? (v * 2 + half * 8) : (16 + (v - 4) * 2 + half * 8);
          const int k    = k0 + kb + lohi;
          const int kc   = clampi(k, C - 1);
          float av = W[(size_t)oc * C + kc];           // always legal
          afrag[e] = (_Float16)((orow < O && k < C) ? av : 0.0f);
        }
      }
      acc[t] = __builtin_amdgcn_wmma_f32_16x16x32_f16(false, afrag, false, bfrag,
                                                      (short)0, acc[t], false, false);
    }
  }
  // ---- epilogue: bias (+ReLU), C/D layout row = i + 8*half ----------
#pragma unroll
  for (int t = 0; t < 4; ++t) {
#pragma unroll
    for (int i = 0; i < 8; ++i) {
      const int o = oT + t * 16 + i + half * 8;
      if (o < O && sOK) {
        float v = acc[t][i] + bias[o];
        if (relu) v = fmaxf(v, 0.0f);
        Yb[(size_t)o * S + scol] = v;
      }
    }
  }
}

// ============================ BatchNorm ==============================
__global__ void pn2_bn_stats(const float* __restrict__ X, float* __restrict__ mean,
                             float* __restrict__ var, int B, int C, long long S) {
  const int c = blockIdx.x;
  __shared__ float ss[256], sq[256];
  float s = 0.f, q = 0.f;
  const long long cnt = (long long)B * S;
  for (long long i = threadIdx.x; i < cnt; i += blockDim.x) {
    const long long bb = i / S, j = i % S;
    const float v = X[((size_t)bb * C + c) * S + j];
    s += v; q += v * v;
  }
  ss[threadIdx.x] = s; sq[threadIdx.x] = q;
  __syncthreads();
  for (int st = 128; st > 0; st >>= 1) {
    if ((int)threadIdx.x < st) {
      ss[threadIdx.x] += ss[threadIdx.x + st];
      sq[threadIdx.x] += sq[threadIdx.x + st];
    }
    __syncthreads();
  }
  if (threadIdx.x == 0) {
    const float inv = 1.0f / (float)cnt;
    const float m = ss[0] * inv;
    mean[c] = m;
    var[c]  = sq[0] * inv - m * m;
  }
}

__global__ void pn2_bn_apply_relu(float* __restrict__ X, const float* __restrict__ mean,
                                  const float* __restrict__ var, const float* __restrict__ g,
                                  const float* __restrict__ beta, int C, long long S,
                                  long long total) {
  const long long t = (long long)blockIdx.x * blockDim.x + threadIdx.x;
  if (t >= total) return;
  const int c = (int)((t / S) % C);
  const float y = (X[t] - mean[c]) * rsqrtf(var[c] + BN_EPS) * g[c] + beta[c];
  X[t] = fmaxf(y, 0.0f);
}

// ===================== center sampling (linspace) =====================
__global__ void pn2_centers(const float* __restrict__ xyz, float* __restrict__ ctr,
                            int B, int P, int M) {
  const long long t = (long long)blockIdx.x * blockDim.x + threadIdx.x;
  if (t >= (long long)B * M) return;
  const int b = (int)(t / M), m = (int)(t % M);
  const int ci = (M > 1) ? (int)((float)m * (float)(P - 1) / (float)(M - 1)) : 0;
#pragma unroll
  for (int c = 0; c < 3; ++c)
    ctr[((size_t)b * M + m) * 3 + c] = xyz[((size_t)b * P + ci) * 3 + c];
}

// ===================== top-32 NN (LDS distance row) ===================
__global__ void pn2_topk32(const float* __restrict__ ctr, const float* __restrict__ pts,
                           int* __restrict__ idx, int M, int P) {
  extern __shared__ float sd[];            // P floats of squared distances
  __shared__ float rv[256]; __shared__ int ri[256];
  const int bm = blockIdx.x;
  const int b = bm / M, m = bm % M;
  const float cx = ctr[((size_t)b * M + m) * 3 + 0];
  const float cy = ctr[((size_t)b * M + m) * 3 + 1];
  const float cz = ctr[((size_t)b * M + m) * 3 + 2];
  for (int n = threadIdx.x; n < P; n += blockDim.x) {
    const float dx = pts[((size_t)b * P + n) * 3 + 0] - cx;
    const float dy = pts[((size_t)b * P + n) * 3 + 1] - cy;
    const float dz = pts[((size_t)b * P + n) * 3 + 2] - cz;
    sd[n] = dx * dx + dy * dy + dz * dz;
  }
  __syncthreads();
  int* out = idx + (size_t)bm * 32;
  for (int it = 0; it < 32; ++it) {
    float best = 3.4e38f; int bi = P - 1;
    for (int n = threadIdx.x; n < P; n += blockDim.x) {
      const float v = sd[n];
      if (v < best || (v == best && n < bi)) { best = v; bi = n; }
    }
    rv[threadIdx.x] = best; ri[threadIdx.x] = bi;
    __syncthreads();
    for (int st = 128; st > 0; st >>= 1) {
      if ((int)threadIdx.x < st) {
        const float ov = rv[threadIdx.x + st]; const int oi = ri[threadIdx.x + st];
        if (ov < rv[threadIdx.x] || (ov == rv[threadIdx.x] && oi < ri[threadIdx.x])) {
          rv[threadIdx.x] = ov; ri[threadIdx.x] = oi;
        }
      }
      __syncthreads();
    }
    if (threadIdx.x == 0) { out[it] = ri[0]; sd[ri[0]] = 3.4e38f; }
    __syncthreads();
  }
}

// ======= group gather: X[b, 0..2, m, k]=local_xyz ; [3..] = feats =====
__global__ void pn2_build_group(const float* __restrict__ xyz, const float* __restrict__ ctr,
                                const float* __restrict__ feats, const int* __restrict__ idx,
                                float* __restrict__ X, int B, int P, int M, int Cf,
                                long long total) {
  const long long t = (long long)blockIdx.x * blockDim.x + threadIdx.x;
  if (t >= total) return;
  const int K = 32;
  const int k = (int)(t % K);
  const int m = (int)((t / K) % M);
  const int c = (int)((t / ((long long)K * M)) % (3 + Cf));
  const int b = (int)(t / ((long long)K * M * (3 + Cf)));
  const int id = idx[((size_t)b * M + m) * K + k];
  float v;
  if (c < 3)
    v = xyz[((size_t)b * P + id) * 3 + c] - ctr[((size_t)b * M + m) * 3 + c];
  else
    v = feats[((size_t)b * Cf + (c - 3)) * P + id];
  X[t] = v;
}

// ======= softmax over nsample + weighted channel sum (one wave/center) =
__global__ __launch_bounds__(32)
void pn2_softmax_wsum(const float* __restrict__ X, const float* __restrict__ wl,
                      float* __restrict__ F, int C, int M) {
  const int bm = blockIdx.x;
  const int b = bm / M, m = bm % M;
  const int k = threadIdx.x & 31;
  float w = wl[((size_t)b * M + m) * 32 + k];
  float mx = w;
#pragma unroll
  for (int o = 16; o > 0; o >>= 1) mx = fmaxf(mx, __shfl_xor(mx, o, 32));
  const float e = __expf(w - mx);
  float s = e;
#pragma unroll
  for (int o = 16; o > 0; o >>= 1) s += __shfl_xor(s, o, 32);
  const float p = e / s;
  const size_t MS = (size_t)M * 32;
  for (int c = 0; c < C; ++c) {
    float v = X[((size_t)b * C + c) * MS + (size_t)m * 32 + k] * p;
#pragma unroll
    for (int o = 16; o > 0; o >>= 1) v += __shfl_xor(v, o, 32);
    if (k == 0) F[((size_t)b * C + c) * M + m] = v;
  }
}

// ===================== 3-NN + inverse-distance weights ================
__global__ void pn2_knn3(const float* __restrict__ a, const float* __restrict__ c2,
                         int* __restrict__ idx, float* __restrict__ w,
                         int B, int N1, int N2) {
  const long long t = (long long)blockIdx.x * blockDim.x + threadIdx.x;
  if (t >= (long long)B * N1) return;
  const int b = (int)(t / N1), n = (int)(t % N1);
  const float px = a[((size_t)b * N1 + n) * 3 + 0];
  const float py = a[((size_t)b * N1 + n) * 3 + 1];
  const float pz = a[((size_t)b * N1 + n) * 3 + 2];
  float d0 = 3.4e38f, d1 = 3.4e38f, d2 = 3.4e38f;
  int i0 = 0, i1 = 0, i2 = 0;
  for (int j = 0; j < N2; ++j) {
    const float dx = c2[((size_t)b * N2 + j) * 3 + 0] - px;
    const float dy = c2[((size_t)b * N2 + j) * 3 + 1] - py;
    const float dz = c2[((size_t)b * N2 + j) * 3 + 2] - pz;
    const float d = dx * dx + dy * dy + dz * dz;
    if (d < d0)      { d2 = d1; i2 = i1; d1 = d0; i1 = i0; d0 = d; i0 = j; }
    else if (d < d1) { d2 = d1; i2 = i1; d1 = d; i1 = j; }
    else if (d < d2) { d2 = d; i2 = j; }
  }
  const float r0 = 1.0f / fmaxf(sqrtf(fmaxf(d0, 1e-12f)), 1e-8f);
  const float r1 = 1.0f / fmaxf(sqrtf(fmaxf(d1, 1e-12f)), 1e-8f);
  const float r2 = 1.0f / fmaxf(sqrtf(fmaxf(d2, 1e-12f)), 1e-8f);
  const float inv = 1.0f / (r0 + r1 + r2);
  const size_t base = (size_t)t * 3;
  idx[base + 0] = i0; idx[base + 1] = i1; idx[base + 2] = i2;
  w[base + 0] = r0 * inv; w[base + 1] = r1 * inv; w[base + 2] = r2 * inv;
}

// ============ interpolation into channels [0, C2) of OUT ==============
__global__ void pn2_interp(const float* __restrict__ F2, const int* __restrict__ idx,
                           const float* __restrict__ w, float* __restrict__ OUT,
                           int C2, int Ccat, int N1, int N2, long long total) {
  const long long t = (long long)blockIdx.x * blockDim.x + threadIdx.x;
  if (t >= total) return;
  const int n = (int)(t % N1);
  const int c = (int)((t / N1) % C2);
  const int b = (int)(t / ((long long)N1 * C2));
  const size_t base = ((size_t)b * N1 + n) * 3;
  const float* f = F2 + ((size_t)b * C2 + c) * N2;
  const float v = w[base + 0] * f[idx[base + 0]] +
                  w[base + 1] * f[idx[base + 1]] +
                  w[base + 2] * f[idx[base + 2]];
  OUT[((size_t)b * Ccat + c) * N1 + n] = v;
}

// ============ copy feats1 into channels [off, off+C1) of OUT ==========
__global__ void pn2_copy_channels(const float* __restrict__ F1, float* __restrict__ OUT,
                                  int C1, int off, int Ccat, int N1, long long total) {
  const long long t = (long long)blockIdx.x * blockDim.x + threadIdx.x;
  if (t >= total) return;
  const int n = (int)(t % N1);
  const int c = (int)((t / N1) % C1);
  const int b = (int)(t / ((long long)N1 * C1));
  OUT[((size_t)b * Ccat + (off + c)) * N1 + n] = F1[((size_t)b * C1 + c) * N1 + n];
}

// ====================== layout transposes =============================
__global__ void pn2_xyz_to_cmajor(const float* __restrict__ xyz, float* __restrict__ T,
                                  int B, int N) {
  const long long t = (long long)blockIdx.x * blockDim.x + threadIdx.x;
  if (t >= (long long)B * 3 * N) return;
  const int n = (int)(t % N);
  const int c = (int)((t / N) % 3);
  const int b = (int)(t / ((long long)N * 3));
  T[t] = xyz[((size_t)b * N + n) * 3 + c];
}

__global__ void pn2_out_transpose(const float* __restrict__ X, float* __restrict__ out,
                                  int B, int Cc, int N) {
  const long long t = (long long)blockIdx.x * blockDim.x + threadIdx.x;
  if (t >= (long long)B * Cc * N) return;
  const int n = (int)(t % N);
  const int c = (int)((t / N) % Cc);
  const int b = (int)(t / ((long long)N * Cc));
  out[((size_t)b * N + n) * Cc + c] = X[t];
}

// =====================================================================
extern "C" void kernel_launch(void* const* d_in, const int* in_sizes, int n_in,
                              void* d_out, int out_size, void* d_ws, size_t ws_size,
                              hipStream_t stream) {
  const int B = 4, N = 8192, M1 = 2048, M2 = 512;

  // ---- locate xyz (unique size) and param base -----------------------
  const float* xyz;
  int pb;
  if (in_sizes[0] == B * N * 3) { xyz = (const float*)d_in[0]; pb = 1; }
  else                          { xyz = (const float*)d_in[n_in - 1]; pb = 0; }

  // pytree order (sorted dict keys): fp1[2], fp2[2], head[2],
  // sa1.local[2], sa1.weight[2], sa2.local[2], sa2.weight[2], spfe[3];
  // per layer: b, beta, g, w
  enum { FP1_0 = 0, FP1_1, FP2_0, FP2_1, HEAD0, HEAD1,
         SA1L0, SA1L1, SA1W0, SA1W1, SA2L0, SA2L1, SA2W0, SA2W1,
         SPFE0, SPFE1, SPFE2 };
  auto Pb = [&](int L) { return (const float*)d_in[pb + L * 4 + 0]; };
  auto Pbeta = [&](int L) { return (const float*)d_in[pb + L * 4 + 1]; };
  auto Pg = [&](int L) { return (const float*)d_in[pb + L * 4 + 2]; };
  auto Pw = [&](int L) { return (const float*)d_in[pb + L * 4 + 3]; };

  // ---- workspace carving (floats) ------------------------------------
  float* ws = (float*)d_ws;
  size_t off = 0;
  auto alloc = [&](size_t nfl) { float* p = ws + off; off += (nfl + 63) & ~(size_t)63; return p; };
  float* R_D  = alloc(67108864);  // big arena: (B,256,65536)/(B,512,16384)/f0u
  float* R_X  = alloc(34340864);  // group-cat / W-branch / concat / head tmp
  float* R_H  = alloc(33554432);  // (B,128,65536) / fp tmps
  float* R_L  = alloc(327680);    // softmax logits / final (B,10,N)
  float* f0   = alloc((size_t)B * 128 * N);
  float* f1   = alloc((size_t)B * 256 * M1);
  float* f2   = alloc((size_t)B * 512 * M2);
  float* f1u  = alloc((size_t)B * 256 * M1);
  float* xyzT = alloc((size_t)B * 3 * N);
  float* ctr1 = alloc((size_t)B * M1 * 3);
  float* ctr2 = alloc((size_t)B * M2 * 3);
  float* stat = alloc(1024);                   // mean[512] + var[512]
  float* wf1  = alloc((size_t)B * M1 * 3);
  float* wf2  = alloc((size_t)B * N * 3);
  int* idx1 = (int*)alloc((size_t)B * M1 * 32);
  int* idx2 = (int*)alloc((size_t)B * M2 * 32);
  int* ixf1 = (int*)alloc((size_t)B * M1 * 3);
  int* ixf2 = (int*)alloc((size_t)B * N * 3);
  if (off * sizeof(float) > ws_size) return;   // insufficient workspace
  float* mean = stat;
  float* var  = stat + 512;

  // ---- launch helpers -------------------------------------------------
  auto gemm = [&](const float* W, const float* X, const float* bias, float* Y,
                  int O, int C, int S, int relu) {
    dim3 grid((S + 15) / 16, (O + 63) / 64, B);
    pn2_gemm_wmma<<<grid, dim3(32), 0, stream>>>(W, X, bias, Y, O, C, S, relu);
  };
  auto bn = [&](float* X, int L, int C, long long S) {
    pn2_bn_stats<<<C, 256, 0, stream>>>(X, mean, var, B, C, S);
    const long long total = (long long)B * C * S;
    pn2_bn_apply_relu<<<(unsigned)((total + 255) / 256), 256, 0, stream>>>(
        X, mean, var, Pg(L), Pbeta(L), C, S, total);
  };
  auto g1d = [&](long long total) { return (unsigned)((total + 255) / 256); };

  // ================= spfe: 3 -> 64 -> 64 -> 128 =======================
  pn2_xyz_to_cmajor<<<g1d((long long)B * 3 * N), 256, 0, stream>>>(xyz, xyzT, B, N);
  gemm(Pw(SPFE0), xyzT, Pb(SPFE0), R_X, 64, 3, N, 0);  bn(R_X, SPFE0, 64, N);
  gemm(Pw(SPFE1), R_X, Pb(SPFE1), R_H, 64, 64, N, 0);  bn(R_H, SPFE1, 64, N);
  gemm(Pw(SPFE2), R_H, Pb(SPFE2), f0, 128, 64, N, 0);  bn(f0, SPFE2, 128, N);

  // ================= sa1: P=8192 -> M=2048 ============================
  pn2_centers<<<g1d((long long)B * M1), 256, 0, stream>>>(xyz, ctr1, B, N, M1);
  pn2_topk32<<<B * M1, 256, N * sizeof(float), stream>>>(ctr1, xyz, idx1, M1, N);
  {
    const long long tot = (long long)B * (3 + 128) * M1 * 32;
    pn2_build_group<<<g1d(tot), 256, 0, stream>>>(xyz, ctr1, f0, idx1, R_X, B, N, M1, 128, tot);
  }
  gemm(Pw(SA1L0), R_X, Pb(SA1L0), R_H, 128, 131, M1 * 32, 0); bn(R_H, SA1L0, 128, (long long)M1 * 32);
  gemm(Pw(SA1L1), R_H, Pb(SA1L1), R_D, 256, 128, M1 * 32, 0); bn(R_D, SA1L1, 256, (long long)M1 * 32);
  gemm(Pw(SA1W0), R_D, Pb(SA1W0), R_X, 128, 256, M1 * 32, 1);   // ReLU fused
  gemm(Pw(SA1W1), R_X, Pb(SA1W1), R_L, 1, 128, M1 * 32, 0);
  pn2_softmax_wsum<<<B * M1, 32, 0, stream>>>(R_D, R_L, f1, 256, M1);

  // ================= sa2: P=2048 -> M=512 =============================
  pn2_centers<<<g1d((long long)B * M2), 256, 0, stream>>>(ctr1, ctr2, B, M1, M2);
  pn2_topk32<<<B * M2, 256, M1 * sizeof(float), stream>>>(ctr2, ctr1, idx2, M2, M1);
  {
    const long long tot = (long long)B * (3 + 256) * M2 * 32;
    pn2_build_group<<<g1d(tot), 256, 0, stream>>>(ctr1, ctr2, f1, idx2, R_X, B, M1, M2, 256, tot);
  }
  gemm(Pw(SA2L0), R_X, Pb(SA2L0), R_H, 128, 259, M2 * 32, 0); bn(R_H, SA2L0, 128, (long long)M2 * 32);
  gemm(Pw(SA2L1), R_H, Pb(SA2L1), R_D, 512, 128, M2 * 32, 0); bn(R_D, SA2L1, 512, (long long)M2 * 32);
  gemm(Pw(SA2W0), R_D, Pb(SA2W0), R_X, 256, 512, M2 * 32, 1);
  gemm(Pw(SA2W1), R_X, Pb(SA2W1), R_L, 1, 256, M2 * 32, 0);
  pn2_softmax_wsum<<<B * M2, 32, 0, stream>>>(R_D, R_L, f2, 512, M2);

  // ================= fp1: (xyz1,f1) <- (xyz2,f2), 768->256->256 =======
  pn2_knn3<<<g1d((long long)B * M1), 256, 0, stream>>>(ctr1, ctr2, ixf1, wf1, B, M1, M2);
  {
    const long long tot = (long long)B * 512 * M1;
    pn2_interp<<<g1d(tot), 256, 0, stream>>>(f2, ixf1, wf1, R_X, 512, 768, M1, M2, tot);
    const long long tot2 = (long long)B * 256 * M1;
    pn2_copy_channels<<<g1d(tot2), 256, 0, stream>>>(f1, R_X, 256, 512, 768, M1, tot2);
  }
  gemm(Pw(FP1_0), R_X, Pb(FP1_0), R_H, 256, 768, M1, 0); bn(R_H, FP1_0, 256, M1);
  gemm(Pw(FP1_1), R_H, Pb(FP1_1), f1u, 256, 256, M1, 0); bn(f1u, FP1_1, 256, M1);

  // ================= fp2: (xyz,f0) <- (xyz1,f1u), 384->256->128 =======
  pn2_knn3<<<g1d((long long)B * N), 256, 0, stream>>>(xyz, ctr1, ixf2, wf2, B, N, M1);
  {
    const long long tot = (long long)B * 256 * N;
    pn2_interp<<<g1d(tot), 256, 0, stream>>>(f1u, ixf2, wf2, R_X, 256, 384, N, M1, tot);
    const long long tot2 = (long long)B * 128 * N;
    pn2_copy_channels<<<g1d(tot2), 256, 0, stream>>>(f0, R_X, 128, 256, 384, N, tot2);
  }
  gemm(Pw(FP2_0), R_X, Pb(FP2_0), R_H, 256, 384, N, 0); bn(R_H, FP2_0, 256, N);
  gemm(Pw(FP2_1), R_H, Pb(FP2_1), R_D, 128, 256, N, 0); bn(R_D, FP2_1, 128, N);

  // ================= head: 128 -> 128 -> 10 ===========================
  gemm(Pw(HEAD0), R_D, Pb(HEAD0), R_X, 128, 128, N, 0); bn(R_X, HEAD0, 128, N);
  gemm(Pw(HEAD1), R_X, Pb(HEAD1), R_L, 10, 128, N, 0);
  pn2_out_transpose<<<g1d((long long)B * 10 * N), 256, 0, stream>>>(R_L, (float*)d_out, B, 10, N);
}
